// DenseGeneralAqt_27066883899833
// MI455X (gfx1250) — compile-verified, matrix operand order checked by reading, not run on hardware
//
#include <hip/hip_runtime.h>

// ---------------------------------------------------------------------------
// DenseGeneralAqt inference GEMM for gfx1250 (MI455X):
//   out[M,N] = (f32 x[M,K]) x (int8 w[K,N] -> bf16) * scale[N]
//   M = B*S = 4096, K = D = 1024, N = F = 4096
//
// Three kernels:
//  1) wt_convert_kernel: w[K][N] int8 -> wt[N][K] bf16 (exact) in d_ws.
//  2) xsplit_kernel:     x f32 -> xhi + xlo bf16 (hi = bf16(x),
//                        lo = bf16(x - hi)) in d_ws. Split-bf16 GEMM
//                        recovers near-f32 accuracy at 2x bf16-WMMA cost.
//  3) main GEMM: conversion-free hot loop. A-hi / A-lo / B tiles staged with
//     global_load_async_to_lds_b128 (ASYNCcnt), double buffered with
//     s_wait_asynccnt + barrier; fragments via ds_load_b128; compute is
//     16x v_wmma_f32_16x16x32_bf16 per wave per K-step.
// ---------------------------------------------------------------------------

typedef __bf16 bf16x16 __attribute__((ext_vector_type(16)));
typedef __bf16 bf16x8  __attribute__((ext_vector_type(8)));
typedef float  f32x8   __attribute__((ext_vector_type(8)));
typedef float  f32x4   __attribute__((ext_vector_type(4)));

#define BM 128
#define BN 128
#define BK 32
#define TLD 40            // bf16 elements per LDS tile row (32 + pad; 80B stride)
#define KDIM 1024
#define NDIM 4096
#define NKSTEP (KDIM / BK)

#if defined(__gfx1250__) && __has_builtin(__builtin_amdgcn_global_load_async_to_lds_b128)
#define USE_ASYNC_LDS 1
#else
#define USE_ASYNC_LDS 0
#endif

// Builtin parameter types (probe-confirmed via hipcc diagnostics): pointer to
// 16-byte int vector; global side AS1, LDS side AS3.
typedef int v4i32 __attribute__((vector_size(16)));
typedef v4i32 __attribute__((address_space(1))) v4i32_g;
typedef v4i32 __attribute__((address_space(3))) v4i32_l;

__device__ inline void copy16_g2lds(void* l, const void* g) {
#if USE_ASYNC_LDS
  __builtin_amdgcn_global_load_async_to_lds_b128(
      (v4i32_g*)(unsigned long long)g,
      (v4i32_l*)(unsigned int)(unsigned long long)l,
      0, 0);
#else
  *(f32x4*)l = *(const f32x4*)g;
#endif
}

__device__ inline void wait_lds_copies() {
#if USE_ASYNC_LDS
#if __has_builtin(__builtin_amdgcn_s_wait_asynccnt)
  __builtin_amdgcn_s_wait_asynccnt(0);
#else
  asm volatile("s_wait_asynccnt 0" ::: "memory");
#endif
#endif
}

__device__ inline f32x8 wmma_bf16(bf16x16 a, bf16x16 b, f32x8 c) {
  // (neg_a, A, neg_b, B, c_mod, C, reuse_a, reuse_b)
  return __builtin_amdgcn_wmma_f32_16x16x32_bf16(false, a, false, b, (short)0, c,
                                                 false, false);
}

union BFrag {
  bf16x16 v;
  bf16x8  h[2];
};

// ---------------------------------------------------------------------------
// Pre-pass 1: wt[n][k] = (bf16)w[k][n]   (64x64 LDS-tiled transpose-convert)
// ---------------------------------------------------------------------------
__global__ __launch_bounds__(256)
void wt_convert_kernel(const signed char* __restrict__ w,
                       __bf16* __restrict__ wt) {
  __shared__ __bf16 t[64 * 72];            // [n][k], 144B row stride (16B mult)
  const int tid = threadIdx.x;
  const int K0  = blockIdx.x * 64;
  const int N0  = blockIdx.y * 64;
  {
    const int k  = tid >> 2;               // 0..63
    const int n0 = (tid & 3) * 16;         // 0,16,32,48
    const signed char* g = w + (size_t)(K0 + k) * NDIM + N0 + n0;
    v4i32 q = *(const v4i32*)g;            // 16 int8, coalesced
    const signed char* p = (const signed char*)&q;
#pragma unroll
    for (int j = 0; j < 16; ++j) {
      t[(n0 + j) * 72 + k] = (__bf16)(float)p[j];
    }
  }
  __syncthreads();
  {
    const int n = tid >> 2;                // 0..63
    const int q = tid & 3;                 // 0..3 -> 16 bf16 chunk
    __bf16* o = wt + (size_t)(N0 + n) * KDIM + K0 + q * 16;
    const __bf16* s = &t[n * 72 + q * 16];
    *(v4i32*)o       = *(const v4i32*)s;        // 16B
    *((v4i32*)o + 1) = *((const v4i32*)s + 1);  // 16B
  }
}

// ---------------------------------------------------------------------------
// Pre-pass 2: xhi = bf16(x), xlo = bf16(x - f32(xhi))   (elementwise)
// ---------------------------------------------------------------------------
__global__ __launch_bounds__(256)
void xsplit_kernel(const float* __restrict__ x,
                   __bf16* __restrict__ xhi,
                   __bf16* __restrict__ xlo) {
  const size_t base = ((size_t)blockIdx.x * 256 + threadIdx.x) * 8;
  f32x4 a = *(const f32x4*)(x + base);
  f32x4 b = *(const f32x4*)(x + base + 4);
  float v[8] = {a[0], a[1], a[2], a[3], b[0], b[1], b[2], b[3]};
  bf16x8 hi, lo;
#pragma unroll
  for (int j = 0; j < 8; ++j) {
    __bf16 h = (__bf16)v[j];
    hi[j] = h;
    lo[j] = (__bf16)(v[j] - (float)h);
  }
  *(bf16x8*)(xhi + base) = hi;
  *(bf16x8*)(xlo + base) = lo;
}

// ---------------------------------------------------------------------------
// Main GEMM: conversion-free split-bf16 WMMA loop
// ---------------------------------------------------------------------------
__global__ __launch_bounds__(256)
void dense_aqt_bf16split_kernel(const __bf16* __restrict__ xhi,  // [M][K]
                                const __bf16* __restrict__ xlo,  // [M][K]
                                const __bf16* __restrict__ wt,   // [N][K]
                                const float* __restrict__ scale,
                                float* __restrict__ out) {
  const int tid  = threadIdx.x;
  const int wave = tid >> 5;        // 8 waves (wave32)
  const int lane = tid & 31;
  const int wm   = wave >> 1;       // 0..3  -> 32-row wave tile
  const int wn   = wave & 1;        // 0..1  -> 64-col wave tile
  const int lgrp = lane >> 4;       // half-wave group (K-group per WMMA layout)
  const int lid  = lane & 15;
  const int row0 = blockIdx.y * BM;
  const int col0 = blockIdx.x * BN;

  __shared__ __bf16 Ah[2][BM * TLD];   // A-hi tile
  __shared__ __bf16 Al[2][BM * TLD];   // A-lo tile
  __shared__ __bf16 Bs[2][BN * TLD];   // B tile
  // 3 * 2 * 128 * 40 * 2B = 60 KB total

  f32x8 acc[2][4] = {};

  // ---- stage one 128x32 bf16 tile (8KB): 2 async b128 per thread -----------
  auto copyTile = [&](__bf16* lds, const __bf16* src, int ks) {
    const int kk = ks * BK;
#pragma unroll
    for (int c = 0; c < 2; ++c) {
      const int id = c * 256 + tid;
      const int r  = id >> 2;          // 0..127
      const int q  = (id & 3) * 8;     // bf16 offset within row
      copy16_g2lds(lds + r * TLD + q, src + (size_t)r * KDIM + kk + q);
    }
  };

  auto stage = [&](int buf, int ks) {
    copyTile(Ah[buf], xhi + (size_t)row0 * KDIM, ks);
    copyTile(Al[buf], xlo + (size_t)row0 * KDIM, ks);
    copyTile(Bs[buf], wt + (size_t)col0 * KDIM, ks);
  };

  // ---- 16x16x32 bf16 fragment: lanes 0-15 take K0-7/K16-23, lanes 16-31
  //      take K8-15/K24-31 (pairs packed per VGPR) ---------------------------
  auto loadFrag = [&](const __bf16* bp) -> bf16x16 {
    BFrag f;
    f.h[0] = *(const bf16x8*)(bp + lgrp * 8);
    f.h[1] = *(const bf16x8*)(bp + 16 + lgrp * 8);
    return f.v;
  };

  // ---- one BK=32 K-step: 16 WMMAs, zero conversion VALU --------------------
  auto compute = [&](int buf) {
    bf16x16 bfrag[4];
#pragma unroll
    for (int nt = 0; nt < 4; ++nt) {
      bfrag[nt] = loadFrag(&Bs[buf][(wn * 64 + nt * 16 + lid) * TLD]);
    }
#pragma unroll
    for (int mt = 0; mt < 2; ++mt) {
      const int ar = (wm * 32 + mt * 16 + lid) * TLD;
      bf16x16 ahi = loadFrag(&Ah[buf][ar]);
      bf16x16 alo = loadFrag(&Al[buf][ar]);
#pragma unroll
      for (int nt = 0; nt < 4; ++nt) {
        acc[mt][nt] = wmma_bf16(ahi, bfrag[nt], acc[mt][nt]);
        acc[mt][nt] = wmma_bf16(alo, bfrag[nt], acc[mt][nt]);
      }
    }
  };

  // ---- double-buffered main loop -------------------------------------------
  stage(0, 0);
#pragma unroll 1
  for (int ks = 0; ks < NKSTEP; ++ks) {
    const int cur = ks & 1;
    wait_lds_copies();     // own async copies done
    __syncthreads();       // all waves' copies visible; prev compute finished
    if (ks + 1 < NKSTEP) {
      stage(cur ^ 1, ks + 1);
    }
    compute(cur);
  }

  // ---- epilogue: per-output-channel scale, then f32 stores -----------------
  // C/D layout: lanes 0-15 -> N=lid, M=v; lanes 16-31 -> N=lid, M=8+v.
#pragma unroll
  for (int mt = 0; mt < 2; ++mt) {
#pragma unroll
    for (int nt = 0; nt < 4; ++nt) {
      const int gn  = col0 + wn * 64 + nt * 16 + lid;
      const float s = scale[gn];
      float* op = out + (size_t)(row0 + wm * 32 + mt * 16 + 8 * lgrp) * NDIM + gn;
#pragma unroll
      for (int v = 0; v < 8; ++v) {
        op[(size_t)v * NDIM] = acc[mt][nt][v] * s;
      }
    }
  }
}

extern "C" void kernel_launch(void* const* d_in, const int* in_sizes, int n_in,
                              void* d_out, int out_size, void* d_ws, size_t ws_size,
                              hipStream_t stream) {
  (void)in_sizes; (void)n_in; (void)ws_size; (void)out_size;
  const float*       x  = (const float*)d_in[0];        // [2,2048,1024] f32
  const signed char* w  = (const signed char*)d_in[1];  // [1024,4096] int8
  const float*       sc = (const float*)d_in[2];        // [1,4096] f32
  float*             o  = (float*)d_out;                // [2,2048,4096] f32

  const int M = 2 * 2048;                               // B*S
  // Workspace layout (24 MB total):
  __bf16* wt  = (__bf16*)d_ws;                          // [4096][1024] (8 MB)
  __bf16* xhi = wt + (size_t)NDIM * KDIM;               // [4096][1024] (8 MB)
  __bf16* xlo = xhi + (size_t)M * KDIM;                 // [4096][1024] (8 MB)

  // Pre-pass 1: int8 weights -> transposed bf16 (exact).
  hipLaunchKernelGGL(wt_convert_kernel, dim3(KDIM / 64, NDIM / 64), dim3(256),
                     0, stream, w, wt);
  // Pre-pass 2: f32 activations -> bf16 hi/lo split.
  hipLaunchKernelGGL(xsplit_kernel, dim3((M * KDIM) / (256 * 8)), dim3(256),
                     0, stream, x, xhi, xlo);
  // Main WMMA GEMM.
  hipLaunchKernelGGL(dense_aqt_bf16split_kernel, dim3(NDIM / BN, M / BM),
                     dim3(256), 0, stream, xhi, xlo, wt, sc, o);
}
